// NGCF_61632780698009
// MI455X (gfx1250) — matrix operand order repaired and checked by reading, not compiled, not run on hardware
//
#include <hip/hip_runtime.h>

typedef __attribute__((ext_vector_type(16))) _Float16 v16h;
typedef __attribute__((ext_vector_type(8)))  float    v8f;

#define DD 64

// ---------------------------------------------------------------------------
// x0 = concat(user_emb, item_emb) written into column-block 0 of out [N, 256]
// ---------------------------------------------------------------------------
__global__ void ngcf_init_x0(const float* __restrict__ user,
                             const float* __restrict__ item,
                             float* __restrict__ out,
                             int n_users, int n_total, int stride) {
    int t = blockIdx.x * blockDim.x + threadIdx.x;
    if (t >= n_total * DD) return;
    int row = t >> 6;
    int d   = t & 63;
    float v = (row < n_users) ? user[row * DD + d]
                              : item[(row - n_users) * DD + d];
    out[(size_t)row * stride + d] = v;
}

// ---------------------------------------------------------------------------
// one-time f32 -> f16 weight conversion (layout preserved: [l][j][d])
// ---------------------------------------------------------------------------
__global__ void ngcf_convert_w(const float* __restrict__ W1,
                               const float* __restrict__ W2,
                               _Float16* __restrict__ W1h,
                               _Float16* __restrict__ W2h,
                               int n) {
    int t = blockIdx.x * blockDim.x + threadIdx.x;
    if (t >= n) return;
    W1h[t] = (_Float16)W1[t];
    W2h[t] = (_Float16)W2[t];
}

// ---------------------------------------------------------------------------
// zero the msg accumulator (float4 per thread)
// ---------------------------------------------------------------------------
__global__ void ngcf_zero(float4* __restrict__ p, int n4) {
    int t = blockIdx.x * blockDim.x + threadIdx.x;
    if (t >= n4) return;
    float4 z; z.x = 0.f; z.y = 0.f; z.z = 0.f; z.w = 0.f;
    p[t] = z;
}

// ---------------------------------------------------------------------------
// COO SpMM scatter: msg[rows[e], :] += vals[e] * x[cols[e], :]
// 16 lanes per edge, float4 per lane, f32 global atomics (gathers hit L2:
// x is 38.4MB << 192MB L2)
// ---------------------------------------------------------------------------
__global__ void ngcf_spmm(const int*   __restrict__ rows,
                          const int*   __restrict__ cols,
                          const float* __restrict__ vals,
                          const float* __restrict__ xcol,   // out + k*64
                          float*       __restrict__ msg,
                          int E, int stride) {
    long long t = (long long)blockIdx.x * blockDim.x + threadIdx.x;
    if (t >= (long long)E * 16) return;
    int e = (int)(t >> 4);
    int g = (int)(t & 15);
    int r = rows[e];
    int c = cols[e];
    float v = vals[e];
    const float4 xv = *(const float4*)(xcol + (size_t)c * stride + g * 4);
    float* mp = msg + (size_t)r * DD + g * 4;
    atomicAdd(mp + 0, v * xv.x);
    atomicAdd(mp + 1, v * xv.y);
    atomicAdd(mp + 2, v * xv.z);
    atomicAdd(mp + 3, v * xv.w);
}

// ---------------------------------------------------------------------------
// Fused layer update: one wave owns a 16-row node tile.
//   h = msg @ W1^T + b1 + (msg*x) @ W2^T + b2 ; leaky_relu(0.2) ; row L2 norm
// 16x16x32 f16 WMMA, f32 accumulate. K=64 -> 2 k-chunks; N=64 -> 4 n-tiles.
// ---------------------------------------------------------------------------
__global__ __launch_bounds__(256)
void ngcf_layer(const float*    __restrict__ msg,
                const float*    __restrict__ xcol,   // out + k*64
                float*          __restrict__ ocol,   // out + (k+1)*64
                const _Float16* __restrict__ W1h,    // [64][64], row j, col d
                const _Float16* __restrict__ W2h,
                const float*    __restrict__ b1,     // [64]
                const float*    __restrict__ b2,     // [64]
                int stride, int n_total) {
    int wave = threadIdx.x >> 5;
    int lane = threadIdx.x & 31;
    int row0 = (blockIdx.x * 8 + wave) * 16;
    if (row0 >= n_total) return;       // whole-wave exit: EXEC stays all-ones
    int m = lane & 15;                 // A row / B,C column within tile
    int h = lane >> 4;                 // lane half selects K/M sub-blocks

    v8f acc[4];
    v8f zero = {0.f, 0.f, 0.f, 0.f, 0.f, 0.f, 0.f, 0.f};
#pragma unroll
    for (int nt = 0; nt < 4; ++nt) acc[nt] = zero;

    int arow = row0 + m;

#pragma unroll
    for (int c = 0; c < 2; ++c) {      // K chunks of 32
        const float* mrow = msg  + (size_t)arow * DD     + c * 32;
        const float* xrow = xcol + (size_t)arow * stride + c * 32;
        // A layout (16-bit 16x32): lane half h -> elems 0..7: K=h*8+i,
        //                          elems 8..15: K=16+h*8+i
        v16h am, ai;
#pragma unroll
        for (int i = 0; i < 8; ++i) {
            float mv  = mrow[h * 8 + i];
            float xv  = xrow[h * 8 + i];
            am[i]     = (_Float16)mv;
            ai[i]     = (_Float16)(mv * xv);
            float mv2 = mrow[16 + h * 8 + i];
            float xv2 = xrow[16 + h * 8 + i];
            am[8 + i] = (_Float16)mv2;
            ai[8 + i] = (_Float16)(mv2 * xv2);
        }
#pragma unroll
        for (int nt = 0; nt < 4; ++nt) {
            // B layout (16-bit 32x16): lane = column n, elems 0..7: K=h*8+i, elems 8..15: K=16+h*8+i.
            // B[k=d][n=j] = W[j][d] -> two runs of 8 f16 (d=h*8+i, 16+h*8+i) from weight row j.
            const _Float16* p1 = W1h + (size_t)(nt * 16 + m) * DD + c * 32;
            const _Float16* p2 = W2h + (size_t)(nt * 16 + m) * DD + c * 32;
            v16h bv1, bv2;
#pragma unroll
            for (int i = 0; i < 8; ++i) { bv1[i] = p1[h * 8 + i]; bv2[i] = p2[h * 8 + i]; bv1[8 + i] = p1[16 + h * 8 + i]; bv2[8 + i] = p2[16 + h * 8 + i]; }
            acc[nt] = __builtin_amdgcn_wmma_f32_16x16x32_f16(
                false, am, false, bv1, (short)0, acc[nt], false, false);
            acc[nt] = __builtin_amdgcn_wmma_f32_16x16x32_f16(
                false, ai, false, bv2, (short)0, acc[nt], false, false);
        }
    }

    // epilogue: bias + leaky_relu + row L2 normalization
    float sumsq[8];
#pragma unroll
    for (int r = 0; r < 8; ++r) sumsq[r] = 0.f;
#pragma unroll
    for (int nt = 0; nt < 4; ++nt) {
        int j = nt * 16 + m;           // all 8 C elems in a lane share column j
        float bias = b1[j] + b2[j];
#pragma unroll
        for (int r = 0; r < 8; ++r) {
            float v = acc[nt][r] + bias;
            v = (v > 0.f) ? v : 0.2f * v;
            acc[nt][r] = v;
            sumsq[r] += v * v;
        }
    }
    // reduce across the 16 lanes of each half (row M = r + 8*h)
#pragma unroll
    for (int r = 0; r < 8; ++r) {
        sumsq[r] += __shfl_xor(sumsq[r], 8, 32);
        sumsq[r] += __shfl_xor(sumsq[r], 4, 32);
        sumsq[r] += __shfl_xor(sumsq[r], 2, 32);
        sumsq[r] += __shfl_xor(sumsq[r], 1, 32);
    }
    float scale[8];
#pragma unroll
    for (int r = 0; r < 8; ++r)
        scale[r] = 1.0f / fmaxf(sqrtf(sumsq[r]), 1e-12f);

#pragma unroll
    for (int nt = 0; nt < 4; ++nt) {
#pragma unroll
        for (int r = 0; r < 8; ++r) {
            int row = row0 + r + 8 * h;
            ocol[(size_t)row * stride + nt * 16 + m] = acc[nt][r] * scale[r];
        }
    }
}

// ---------------------------------------------------------------------------
extern "C" void kernel_launch(void* const* d_in, const int* in_sizes, int n_in,
                              void* d_out, int out_size, void* d_ws, size_t ws_size,
                              hipStream_t stream) {
    const float* user = (const float*)d_in[0];
    const float* item = (const float*)d_in[1];
    const float* W1   = (const float*)d_in[2];
    const float* b1   = (const float*)d_in[3];
    const float* W2   = (const float*)d_in[4];
    const float* b2   = (const float*)d_in[5];
    const float* vals = (const float*)d_in[6];
    const int*   rows = (const int*)d_in[7];
    const int*   cols = (const int*)d_in[8];
    float* out = (float*)d_out;

    const int n_users = in_sizes[0] / DD;
    const int n_items = in_sizes[1] / DD;
    const int N       = n_users + n_items;          // 150000 (multiple of 16)
    const int L       = in_sizes[2] / (DD * DD);    // 3
    const int E       = in_sizes[6];                // 4.8M
    const int stride  = DD * (L + 1);               // 256

    // workspace: msg [N,64] f32, then f16 weights
    float* msg = (float*)d_ws;
    size_t msg_bytes = (size_t)N * DD * sizeof(float);
    size_t woff = (msg_bytes + 255) & ~(size_t)255;
    _Float16* W1h = (_Float16*)((char*)d_ws + woff);
    _Float16* W2h = W1h + (size_t)L * DD * DD;

    // x0 into column block 0 of out
    {
        int total = N * DD;
        ngcf_init_x0<<<(total + 255) / 256, 256, 0, stream>>>(
            user, item, out, n_users, N, stride);
    }
    // convert weights to f16
    {
        int total = L * DD * DD;
        ngcf_convert_w<<<(total + 255) / 256, 256, 0, stream>>>(
            W1, W2, W1h, W2h, total);
    }

    for (int k = 0; k < L; ++k) {
        const float* xcol = out + (size_t)k * DD;
        float*       ocol = out + (size_t)(k + 1) * DD;

        int n4 = N * DD / 4;
        ngcf_zero<<<(n4 + 255) / 256, 256, 0, stream>>>((float4*)msg, n4);

        long long tthreads = (long long)E * 16;
        int sblocks = (int)((tthreads + 255) / 256);
        ngcf_spmm<<<sblocks, 256, 0, stream>>>(rows, cols, vals, xcol, msg, E, stride);

        int tiles = N / 16;                // one wave per 16-row tile
        ngcf_layer<<<(tiles + 7) / 8, 256, 0, stream>>>(
            msg, xcol, ocol,
            W1h + (size_t)k * DD * DD, W2h + (size_t)k * DD * DD,
            b1 + (size_t)k * DD, b2 + (size_t)k * DD,
            stride, N);
    }
}
